// PointNet_FP_Module_1967095021877
// MI455X (gfx1250) — matrix operand order repaired and checked by reading, not compiled, not run on hardware
//
#include <hip/hip_runtime.h>
#include <hip/hip_bf16.h>
#include <cstdint>

// ---------------- problem constants (from reference) ----------------
#define BB   16
#define NN   4096
#define MM   1024
#define C2   256
#define C1   128
#define IN0  384     // C2 + C1
#define O1   256
#define O2   256

typedef _Float16 half_t;
typedef __attribute__((ext_vector_type(16))) _Float16 v16h;
typedef __attribute__((ext_vector_type(8)))  float    v8f;

// Load a 16-half WMMA fragment as two contiguous 16B chunks.
__device__ __forceinline__ v16h load_frag2(const half_t* p0, const half_t* p1) {
    union { uint4 u[2]; v16h v; } r;
    r.u[0] = *reinterpret_cast<const uint4*>(p0);
    r.u[1] = *reinterpret_cast<const uint4*>(p1);
    return r.v;
}

// ---------------- kernel: f32 -> f16 convert (weights) ----------------
__global__ void k_cvt_f16(const float* __restrict__ src, half_t* __restrict__ dst, int n) {
    int i = blockIdx.x * blockDim.x + threadIdx.x;
    if (i < n) dst[i] = (half_t)src[i];
}

// ---------------- kernel: transpose points2 (B,C2,M) -> (B,M,C2) ----------------
__global__ void k_transpose_p2(const float* __restrict__ p2, float* __restrict__ p2t) {
    __shared__ float tile[32][33];
    int b  = blockIdx.z;
    int m0 = blockIdx.x * 32;
    int c0 = blockIdx.y * 32;
    int tx = threadIdx.x;           // 0..31 (along M on read)
    int ty = threadIdx.y;           // 0..7
    #pragma unroll
    for (int i = 0; i < 4; ++i) {
        int c = c0 + ty + i * 8;
        tile[ty + i * 8][tx] = p2[((size_t)b * C2 + c) * MM + (m0 + tx)];
    }
    __syncthreads();
    #pragma unroll
    for (int i = 0; i < 4; ++i) {
        int m = m0 + ty + i * 8;
        p2t[((size_t)b * MM + m) * C2 + (c0 + tx)] = tile[tx][ty + i * 8];
    }
}

// ---------------- kernel: brute-force 3-NN + inverse-distance weights ----------------
// grid (N/256, B), block 256. xyz2 for batch b staged in LDS.
__global__ void k_three_nn(const float* __restrict__ xyz1, const float* __restrict__ xyz2,
                           int* __restrict__ idx, float* __restrict__ wgt) {
    __shared__ float s2x[MM], s2y[MM], s2z[MM];
    int b = blockIdx.y;
    int t = threadIdx.x;
    for (int i = t; i < MM; i += 256) {
        s2x[i] = xyz2[((size_t)b * 3 + 0) * MM + i];
        s2y[i] = xyz2[((size_t)b * 3 + 1) * MM + i];
        s2z[i] = xyz2[((size_t)b * 3 + 2) * MM + i];
    }
    __syncthreads();

    int n = blockIdx.x * 256 + t;
    float x = xyz1[((size_t)b * 3 + 0) * NN + n];
    float y = xyz1[((size_t)b * 3 + 1) * NN + n];
    float z = xyz1[((size_t)b * 3 + 2) * NN + n];

    float d0 = 3.4e38f, d1 = 3.4e38f, d2 = 3.4e38f;
    int   i0 = 0, i1 = 0, i2 = 0;
    for (int m = 0; m < MM; ++m) {
        float dx = x - s2x[m], dy = y - s2y[m], dz = z - s2z[m];
        float d = dx * dx + dy * dy + dz * dz;
        if (d < d0)      { d2 = d1; i2 = i1; d1 = d0; i1 = i0; d0 = d; i0 = m; }
        else if (d < d1) { d2 = d1; i2 = i1; d1 = d;  i1 = m; }
        else if (d < d2) { d2 = d;  i2 = m; }
    }
    float s0 = sqrtf(fmaxf(d0, 1e-20f)); s0 = fmaxf(s0, 1e-10f);
    float s1 = sqrtf(fmaxf(d1, 1e-20f)); s1 = fmaxf(s1, 1e-10f);
    float s2 = sqrtf(fmaxf(d2, 1e-20f)); s2 = fmaxf(s2, 1e-10f);
    float r0 = 1.0f / s0, r1 = 1.0f / s1, r2 = 1.0f / s2;
    float rs = 1.0f / (r0 + r1 + r2);
    size_t base = ((size_t)b * NN + n) * 3;
    idx[base + 0] = i0; idx[base + 1] = i1; idx[base + 2] = i2;
    wgt[base + 0] = r0 * rs; wgt[base + 1] = r1 * rs; wgt[base + 2] = r2 * rs;
}

// ---------------- kernel: gather + interpolate + concat -> Xcat (B,N,384) f16 ----------------
// grid (N, B), block 384 (12 waves). c<256: interp from p2t; c>=256: copy points1.
__global__ void k_interp_concat(const float* __restrict__ p2t, const float* __restrict__ pts1,
                                const int* __restrict__ idx, const float* __restrict__ wgt,
                                half_t* __restrict__ xcat) {
    int b = blockIdx.y, n = blockIdx.x, c = threadIdx.x;
    size_t nb = (size_t)b * NN + n;
    int   i0 = idx[nb * 3 + 0], i1 = idx[nb * 3 + 1], i2 = idx[nb * 3 + 2];
    float w0 = wgt[nb * 3 + 0], w1 = wgt[nb * 3 + 1], w2 = wgt[nb * 3 + 2];
    float v;
    if (c < C2) {
        v = w0 * p2t[((size_t)b * MM + i0) * C2 + c]
          + w1 * p2t[((size_t)b * MM + i1) * C2 + c]
          + w2 * p2t[((size_t)b * MM + i2) * C2 + c];
    } else {
        v = pts1[((size_t)b * C1 + (c - C2)) * NN + n];
    }
    xcat[nb * IN0 + c] = (half_t)v;
}

// ---------------- WMMA GEMM: Y = relu(W (OxK) * X (B,N,K)^T + bias) ----------------
// grid (N/64, O/128, B), block 128 (4 waves). Each wave: 32(O) x 64(N) tile:
// 8 independent f32 accumulators, 2 A fragments reused across 4 B fragments per
// K-step => 12 b128 loads per 8 v_wmma (1.5 loads/WMMA) + lots of ILP to hide
// load latency within a wave.
// FINAL=false: store f16 to outh (B,N,O) n-major (feeds next GEMM's B-layout).
// FINAL=true : store f32 to outf (B,O,N) o-major (matches reference output).
template <int KDIM, bool FINAL>
__global__ void k_gemm_wmma(const half_t* __restrict__ W, const half_t* __restrict__ X,
                            const float* __restrict__ bias,
                            half_t* __restrict__ outh, float* __restrict__ outf, int Odim) {
    const int lane  = threadIdx.x & 31;
    const int wave  = threadIdx.x >> 5;
    const int b     = blockIdx.z;
    const int nBase = blockIdx.x * 64;
    const int waveO = blockIdx.y * 128 + wave * 32;

    const int l16  = lane & 15;     // A row / B-D column within tile
    const int lhi  = lane >> 4;     // 0/1 half selector

    v8f acc[2][4];
    #pragma unroll
    for (int g = 0; g < 2; ++g)
        #pragma unroll
        for (int t = 0; t < 4; ++t)
            acc[g][t] = (v8f){0.f, 0.f, 0.f, 0.f, 0.f, 0.f, 0.f, 0.f};

    const half_t* arow0 = W + (size_t)(waveO + l16) * KDIM;        // rows m .. m of group 0
    const half_t* arow1 = arow0 + (size_t)16 * KDIM;               // rows m+16 (group 1)
    const half_t* xrow  = X + ((size_t)b * NN + nBase + l16) * KDIM + lhi * 16;

    for (int k0 = 0; k0 < KDIM; k0 += 32) {
        // A fragments: lane row l16, K = k0+lhi*16+{0..15}
        v16h a0 = load_frag2(arow0 + k0 + lhi * 16, arow0 + k0 + lhi * 16 + 8);
        v16h a1 = load_frag2(arow1 + k0 + lhi * 16, arow1 + k0 + lhi * 16 + 8);
        if (k0 + 32 < KDIM) __builtin_prefetch(xrow + k0 + 32, 0, 1);   // global_prefetch
        #pragma unroll
        for (int t = 0; t < 4; ++t) {
            // B fragment: column n = nBase + t*16 + l16, K = k0 + lhi*16 + {0..15}
            const half_t* xc = xrow + (size_t)t * 16 * KDIM + k0;
            v16h bf = load_frag2(xc, xc + 8);
            acc[0][t] = __builtin_amdgcn_wmma_f32_16x16x32_f16(
                false, a0, false, bf, (short)0, acc[0][t], false, false);
            acc[1][t] = __builtin_amdgcn_wmma_f32_16x16x32_f16(
                false, a1, false, bf, (short)0, acc[1][t], false, false);
        }
    }

    // Epilogue: in each 16-row group g, lane holds rows m = g*16 + lhi*8 + r,
    // i.e. 8 contiguous O channels -> single packed b128 store on the f16 path.
    #pragma unroll
    for (int g = 0; g < 2; ++g) {
        const int ob = waveO + g * 16 + lhi * 8;
        float bv[8];
        {
            const float4 b0 = *reinterpret_cast<const float4*>(bias + ob);
            const float4 b1 = *reinterpret_cast<const float4*>(bias + ob + 4);
            bv[0] = b0.x; bv[1] = b0.y; bv[2] = b0.z; bv[3] = b0.w;
            bv[4] = b1.x; bv[5] = b1.y; bv[6] = b1.z; bv[7] = b1.w;
        }
        #pragma unroll
        for (int t = 0; t < 4; ++t) {
            const int n = nBase + t * 16 + l16;
            if (FINAL) {
                float* o = outf + ((size_t)b * Odim + ob) * NN + n;
                #pragma unroll
                for (int r = 0; r < 8; ++r)
                    o[(size_t)r * NN] = fmaxf(acc[g][t][r] + bv[r], 0.0f);
            } else {
                union { half_t h[8]; uint4 u; } pk;
                #pragma unroll
                for (int r = 0; r < 8; ++r)
                    pk.h[r] = (half_t)fmaxf(acc[g][t][r] + bv[r], 0.0f);
                *reinterpret_cast<uint4*>(outh + ((size_t)b * NN + n) * Odim + ob) = pk.u;
            }
        }
    }
}

// ---------------- launch ----------------
extern "C" void kernel_launch(void* const* d_in, const int* in_sizes, int n_in,
                              void* d_out, int out_size, void* d_ws, size_t ws_size,
                              hipStream_t stream) {
    (void)in_sizes; (void)n_in; (void)out_size; (void)ws_size;
    const float* xyz1 = (const float*)d_in[0];
    const float* xyz2 = (const float*)d_in[1];
    const float* pts1 = (const float*)d_in[2];
    const float* pts2 = (const float*)d_in[3];
    const float* W1   = (const float*)d_in[4];
    const float* b1   = (const float*)d_in[5];
    const float* W2   = (const float*)d_in[6];
    const float* b2   = (const float*)d_in[7];
    float* out = (float*)d_out;

    // workspace carve-up (all 256B-aligned chunks)
    uint8_t* ws = (uint8_t*)d_ws;
    size_t off = 0;
    auto take = [&](size_t bytes) { void* p = ws + off; off += (bytes + 255) & ~(size_t)255; return p; };
    float*  p2t  = (float*) take((size_t)BB * MM * C2 * 4);        // 16.8 MB
    int*    idx  = (int*)   take((size_t)BB * NN * 3 * 4);
    float*  wgt  = (float*) take((size_t)BB * NN * 3 * 4);
    half_t* W1h  = (half_t*)take((size_t)O1 * IN0 * 2);
    half_t* W2h  = (half_t*)take((size_t)O2 * O1 * 2);
    half_t* xcat = (half_t*)take((size_t)BB * NN * IN0 * 2);       // 50.3 MB
    half_t* y1   = (half_t*)take((size_t)BB * NN * O1 * 2);        // 33.5 MB

    // 1) weights -> f16
    k_cvt_f16<<<(O1 * IN0 + 255) / 256, 256, 0, stream>>>(W1, W1h, O1 * IN0);
    k_cvt_f16<<<(O2 * O1 + 255) / 256, 256, 0, stream>>>(W2, W2h, O2 * O1);

    // 2) transpose points2 for coalesced gathers
    k_transpose_p2<<<dim3(MM / 32, C2 / 32, BB), dim3(32, 8), 0, stream>>>(pts2, p2t);

    // 3) three_nn + weights
    k_three_nn<<<dim3(NN / 256, BB), 256, 0, stream>>>(xyz1, xyz2, idx, wgt);

    // 4) interpolate + concat -> Xcat (B,N,384) f16
    k_interp_concat<<<dim3(NN, BB), IN0, 0, stream>>>(p2t, pts1, idx, wgt, xcat);

    // 5) GEMM1: relu(W1 * Xcat + b1) -> Y1 (B,N,256) f16
    k_gemm_wmma<IN0, false><<<dim3(NN / 64, O1 / 128, BB), 128, 0, stream>>>(
        W1h, xcat, b1, y1, nullptr, O1);

    // 6) GEMM2: relu(W2 * Y1 + b2) -> out (B,256,N) f32
    k_gemm_wmma<O1, true><<<dim3(NN / 64, O2 / 128, BB), 128, 0, stream>>>(
        W2h, y1, b2, nullptr, out, O2);
}
